// MultiHeadAttentionBlock_39728447488169
// MI455X (gfx1250) — matrix operand order repaired and checked by reading, not compiled, not run on hardware
//
#include <hip/hip_runtime.h>
#include <stdint.h>

// ---------------------------------------------------------------------------
// MultiHeadAttentionBlock for MI455X (gfx1250, wave32, WMMA).
// bf16 operands, f32 accumulate, flash-style fused attention (no SxS tensor).
// Round 2: async-to-LDS A-panel staging in GEMMs, 16x32 macro-tiles,
//          V stored transposed per head so P*V loads are contiguous.
// ---------------------------------------------------------------------------

#define D_MODEL   1024
#define NUM_HEADS 16
#define HEAD_DIM  64
#define BATCH     2
#define SEQ       2048
#define NROWS     (BATCH * SEQ)   // 4096 token rows

typedef __bf16 bf16;
typedef __attribute__((ext_vector_type(8)))  __bf16 v8bf;
typedef __attribute__((ext_vector_type(16))) __bf16 v16bf;
typedef __attribute__((ext_vector_type(8)))  float  v8f;

static __device__ __forceinline__ v8f vzero8f() {
  v8f z;
#pragma unroll
  for (int i = 0; i < 8; ++i) z[i] = 0.0f;
  return z;
}

// D(16x16 f32) += A(16x32 bf16) * B(32x16 bf16)
static __device__ __forceinline__ v8f wmma_bf16(v16bf a, v16bf b, v8f c) {
  return __builtin_amdgcn_wmma_f32_16x16x32_bf16(false, a, false, b,
                                                 (short)0, c, false, false);
}

// A fragment (16x32) from row-major [.. x rs] at (m0, k0).
// lane&15 = row; lane>>4 selects K-chunks {0..7,16..23} vs {8..15,24..31}.
static __device__ __forceinline__ v16bf load_a_frag(const bf16* __restrict__ base,
                                                    int rs, int m0, int k0, int lane) {
  const int m   = m0 + (lane & 15);
  const int sel = lane >> 4;
  const bf16* p = base + (size_t)m * rs + k0 + sel * 8;
  v8bf lo = *(const v8bf*)(p);
  v8bf hi = *(const v8bf*)(p + 16);
  v16bf a;
#pragma unroll
  for (int i = 0; i < 8; ++i) { a[i] = lo[i]; a[8 + i] = hi[i]; }
  return a;
}

// B fragment (32x16) with B[k][n] = src[n0+n][k0+k], src row-major [N x rs]:
// each lane's column is one contiguous row of src. lane>>4 selects K-chunks {0..7,16..23} vs {8..15,24..31}.
static __device__ __forceinline__ v16bf load_b_frag_T(const bf16* __restrict__ base,
                                                      int rs, int n0, int k0, int lane) {
  const int n   = n0 + (lane & 15);
  const int sel = lane >> 4;
  const bf16* p = base + (size_t)n * rs + k0 + sel * 8;
  v8bf x0 = *(const v8bf*)(p);
  v8bf x1 = *(const v8bf*)(p + 16);
  v16bf b;
#pragma unroll
  for (int i = 0; i < 8; ++i) { b[i] = x0[i]; b[8 + i] = x1[i]; }
  return b;
}

// ---------------------------------------------------------------------------
// Kernel 1: f32 -> bf16 conversion (grid-stride).
// ---------------------------------------------------------------------------
__global__ void cvt_f32_bf16(const float* __restrict__ src, bf16* __restrict__ dst, int n) {
  int i = blockIdx.x * blockDim.x + threadIdx.x;
  const int stride = gridDim.x * blockDim.x;
  for (; i < n; i += stride) dst[i] = (bf16)src[i];
}

// ---------------------------------------------------------------------------
// Cooperative async staging: copy the contiguous 16x1024 bf16 A panel (32KB)
// into LDS with GLOBAL_LOAD_ASYNC_TO_LDS_B128 (ASYNCcnt), then barrier.
// 256 threads x 8 chunks x 16B = 32KB.
// ---------------------------------------------------------------------------
static __device__ __forceinline__ void stage_a_panel(bf16* lds, const bf16* gsrc,
                                                     int tid) {
  const uint32_t lbase = (uint32_t)(uintptr_t)lds + (uint32_t)tid * 16u;
  const uint64_t gbase = (uint64_t)(uintptr_t)gsrc + (uint64_t)tid * 16u;
#pragma unroll
  for (int ci = 0; ci < 8; ++ci) {
    uint32_t l = lbase + (uint32_t)ci * 4096u;
    uint64_t g = gbase + (uint64_t)ci * 4096u;
    asm volatile("global_load_async_to_lds_b128 %0, %1, off"
                 :: "v"(l), "v"(g) : "memory");
  }
  asm volatile("s_wait_asynccnt 0x0" ::: "memory");
  __syncthreads();
}

// A fragment (16x32) out of the LDS-resident panel [16 x D_MODEL].
static __device__ __forceinline__ v16bf lds_a_frag(const bf16* lds, int k0, int lane) {
  const int m   = lane & 15;
  const int sel = lane >> 4;
  const bf16* r = lds + m * D_MODEL + k0 + sel * 8;
  v16bf a;
#pragma unroll
  for (int i = 0; i < 8; ++i) { a[i] = r[i]; a[8 + i] = r[16 + i]; }
  return a;
}

// ---------------------------------------------------------------------------
// Kernel 2: fused QKV projections. gridDim.z selects {Q,K,V}.
// Y = X @ W^T + b. Q,K stored bf16 [B,H,S,dk]; V stored transposed [B,H,dk,S].
// Per block: one 16-row panel staged in LDS; per wave: 16x32 output macro-tile.
// 8192 waves per projection -> 1024 blocks of 8 waves.
// ---------------------------------------------------------------------------
__global__ void __launch_bounds__(256)
qkv_proj(const bf16* __restrict__ qx, const bf16* __restrict__ kx,
         const bf16* __restrict__ vx,
         const bf16* __restrict__ wq, const bf16* __restrict__ wk,
         const bf16* __restrict__ wv,
         const float* __restrict__ bq, const float* __restrict__ bk,
         const float* __restrict__ bv,
         bf16* __restrict__ Qp, bf16* __restrict__ Kp, bf16* __restrict__ Vp) {
  __shared__ __attribute__((aligned(16))) bf16 Atile[16 * D_MODEL];  // 32 KB

  const int lane = threadIdx.x & 31;
  const int id   = blockIdx.x * 8 + (threadIdx.x >> 5);  // [0, 8192)
  const int mtile = id >> 5;          // 256 row tiles of 16 (uniform per block)
  const int n0    = (id & 31) * 32;   // 32 col macro-tiles of 32

  const bf16* X; const bf16* W; const float* bias; bf16* dst;
  if (blockIdx.z == 0)      { X = qx; W = wq; bias = bq; dst = Qp; }
  else if (blockIdx.z == 1) { X = kx; W = wk; bias = bk; dst = Kp; }
  else                      { X = vx; W = wv; bias = bv; dst = Vp; }
  const bool vtrans = (blockIdx.z == 2);

  stage_a_panel(Atile, X + (size_t)mtile * 16 * D_MODEL, threadIdx.x);

  v8f c0 = vzero8f(), c1 = vzero8f();
#pragma unroll 2
  for (int kk = 0; kk < D_MODEL; kk += 32) {
    v16bf a  = lds_a_frag(Atile, kk, lane);
    v16bf b0 = load_b_frag_T(W, D_MODEL, n0,      kk, lane);  // W rows = W^T cols
    v16bf b1 = load_b_frag_T(W, D_MODEL, n0 + 16, kk, lane);
    c0 = wmma_bf16(a, b0, c0);
    c1 = wmma_bf16(a, b1, c1);
  }

  const int nn  = lane & 15;
  const int sel = lane >> 4;
#pragma unroll
  for (int t = 0; t < 2; ++t) {
    v8f& c = t ? c1 : c0;
    const int col = n0 + t * 16 + nn;
    const int h = col >> 6;         // head
    const int d = col & 63;         // dim within head
    const float bval = bias[col];
#pragma unroll
    for (int j = 0; j < 8; ++j) {
      const int r  = mtile * 16 + j + 8 * sel;    // token row
      const int bb = r >> 11;                     // / SEQ
      const int s  = r & (SEQ - 1);
      const bf16 val = (bf16)(c[j] + bval);
      if (vtrans)
        dst[(((size_t)bb * NUM_HEADS + h) * HEAD_DIM + d) * SEQ + s] = val;
      else
        dst[(((size_t)bb * NUM_HEADS + h) * SEQ + s) * HEAD_DIM + d] = val;
    }
  }
}

// ---------------------------------------------------------------------------
// Kernel 3: causal flash attention. One wave per (b, h, 16-query tile).
// 32-key blocks: 4 WMMAs QK^T -> online softmax (f32) -> P via LDS -> 4 WMMAs PV.
// V is [B,H,dk,S] so PV B-fragments are contiguous 16B loads.
// Output bf16 [B, S, D_MODEL].
// ---------------------------------------------------------------------------
__global__ void __launch_bounds__(128)
flash_attn(const bf16* __restrict__ Qp, const bf16* __restrict__ Kp,
           const bf16* __restrict__ Vp, bf16* __restrict__ Xa) {
  __shared__ __attribute__((aligned(16))) bf16 plds[4][16 * 32];

  const int lane = threadIdx.x & 31;
  const int wid  = threadIdx.x >> 5;
  const int tid  = blockIdx.x * 4 + wid;   // [0, 4096) = B*H*(S/16)
  const int qt = tid & 127;
  const int h  = (tid >> 7) & (NUM_HEADS - 1);
  const int bb = tid >> 11;
  const int q0 = qt * 16;

  const bf16* Qh = Qp + ((size_t)(bb * NUM_HEADS + h)) * SEQ * HEAD_DIM;  // [S,dk]
  const bf16* Kh = Kp + ((size_t)(bb * NUM_HEADS + h)) * SEQ * HEAD_DIM;  // [S,dk]
  const bf16* Vh = Vp + ((size_t)(bb * NUM_HEADS + h)) * HEAD_DIM * SEQ;  // [dk,S]

  const int n   = lane & 15;
  const int sel = lane >> 4;

  // Q A-fragments (rows q0..q0+15, d = 0..31 and 32..63), reused all blocks.
  const v16bf aq0 = load_a_frag(Qh, HEAD_DIM, q0, 0, lane);
  const v16bf aq1 = load_a_frag(Qh, HEAD_DIM, q0, 32, lane);

  v8f o[4];
#pragma unroll
  for (int t = 0; t < 4; ++t) o[t] = vzero8f();
  float m_run[8], l_run[8];
#pragma unroll
  for (int j = 0; j < 8; ++j) { m_run[j] = -3.0e38f; l_run[j] = 0.0f; }

  const int nkb = ((q0 + 15) >> 5) + 1;    // causal: key blocks kb0 <= q0+15
  for (int ib = 0; ib < nkb; ++ib) {
    const int kb0 = ib * 32;

    // scores S[16q x 32k] as two 16x16 f32 fragments
    v8f s0 = vzero8f(), s1 = vzero8f();
    s0 = wmma_bf16(aq0, load_b_frag_T(Kh, HEAD_DIM, kb0,      0,  lane), s0);
    s0 = wmma_bf16(aq1, load_b_frag_T(Kh, HEAD_DIM, kb0,      32, lane), s0);
    s1 = wmma_bf16(aq0, load_b_frag_T(Kh, HEAD_DIM, kb0 + 16, 0,  lane), s1);
    s1 = wmma_bf16(aq1, load_b_frag_T(Kh, HEAD_DIM, kb0 + 16, 32, lane), s1);

    // scale + causal mask (this lane owns column n; rows j + 8*sel)
    float p0[8], p1[8];
#pragma unroll
    for (int j = 0; j < 8; ++j) {
      const int qrow = q0 + j + 8 * sel;
      float v0 = s0[j] * 0.125f;   // 1/sqrt(64)
      float v1 = s1[j] * 0.125f;
      if (kb0 + n > qrow)      v0 = -3.0e38f;
      if (kb0 + 16 + n > qrow) v1 = -3.0e38f;
      p0[j] = v0; p1[j] = v1;
    }

    // row max across 16 columns (lanes within each 16-lane half)
    float rm[8];
#pragma unroll
    for (int j = 0; j < 8; ++j) rm[j] = fmaxf(p0[j], p1[j]);
#pragma unroll
    for (int msk = 1; msk <= 8; msk <<= 1)
#pragma unroll
      for (int j = 0; j < 8; ++j) rm[j] = fmaxf(rm[j], __shfl_xor(rm[j], msk, 32));

    float corr[8], rsum[8];
#pragma unroll
    for (int j = 0; j < 8; ++j) {
      const float mn = fmaxf(m_run[j], rm[j]);
      corr[j]  = __expf(m_run[j] - mn);
      m_run[j] = mn;
      p0[j] = __expf(p0[j] - mn);
      p1[j] = __expf(p1[j] - mn);
      rsum[j] = p0[j] + p1[j];
    }
#pragma unroll
    for (int msk = 1; msk <= 8; msk <<= 1)
#pragma unroll
      for (int j = 0; j < 8; ++j) rsum[j] += __shfl_xor(rsum[j], msk, 32);
#pragma unroll
    for (int j = 0; j < 8; ++j) l_run[j] = l_run[j] * corr[j] + rsum[j];

    // rescale running output
#pragma unroll
    for (int t = 0; t < 4; ++t)
#pragma unroll
      for (int j = 0; j < 8; ++j) o[t][j] *= corr[j];

    // stage P (16x32) row-major in LDS, C-layout scatter -> A-layout gather
#pragma unroll
    for (int j = 0; j < 8; ++j) {
      const int r = j + 8 * sel;
      plds[wid][r * 32 + n]      = (bf16)p0[j];
      plds[wid][r * 32 + 16 + n] = (bf16)p1[j];
    }
    asm volatile("s_wait_dscnt 0x0" ::: "memory");  // cross-lane LDS W->R
    v16bf ap;
    {
      const bf16* row = &plds[wid][(lane & 15) * 32 + sel * 8];
#pragma unroll
      for (int i = 0; i < 8; ++i) { ap[i] = row[i]; ap[8 + i] = row[16 + i]; }
    }

    // O[16 x 64] += P[16 x 32] * V[32 x 64]; Vt rows = head dims -> contiguous
#pragma unroll
    for (int t = 0; t < 4; ++t)
      o[t] = wmma_bf16(ap, load_b_frag_T(Vh, SEQ, t * 16, kb0, lane), o[t]);
  }

  // epilogue: divide by softmax denom, store bf16 [B, S, D_MODEL]
  float rl[8];
#pragma unroll
  for (int j = 0; j < 8; ++j) rl[j] = 1.0f / l_run[j];
#pragma unroll
  for (int t = 0; t < 4; ++t) {
    const int col = h * HEAD_DIM + t * 16 + n;
#pragma unroll
    for (int j = 0; j < 8; ++j) {
      const int r = q0 + j + 8 * sel;
      Xa[((size_t)bb * SEQ + r) * D_MODEL + col] = (bf16)(o[t][j] * rl[j]);
    }
  }
}

// ---------------------------------------------------------------------------
// Kernel 4: output projection (same tiling as qkv_proj), f32 result to d_out.
// ---------------------------------------------------------------------------
__global__ void __launch_bounds__(256)
out_proj(const bf16* __restrict__ X, const bf16* __restrict__ Wo,
         const float* __restrict__ bo, float* __restrict__ out) {
  __shared__ __attribute__((aligned(16))) bf16 Atile[16 * D_MODEL];  // 32 KB

  const int lane = threadIdx.x & 31;
  const int id   = blockIdx.x * 8 + (threadIdx.x >> 5);  // [0, 8192)
  const int mtile = id >> 5;
  const int n0    = (id & 31) * 32;

  stage_a_panel(Atile, X + (size_t)mtile * 16 * D_MODEL, threadIdx.x);

  v8f c0 = vzero8f(), c1 = vzero8f();
#pragma unroll 2
  for (int kk = 0; kk < D_MODEL; kk += 32) {
    v16bf a  = lds_a_frag(Atile, kk, lane);
    v16bf b0 = load_b_frag_T(Wo, D_MODEL, n0,      kk, lane);
    v16bf b1 = load_b_frag_T(Wo, D_MODEL, n0 + 16, kk, lane);
    c0 = wmma_bf16(a, b0, c0);
    c1 = wmma_bf16(a, b1, c1);
  }

  const int nn  = lane & 15;
  const int sel = lane >> 4;
#pragma unroll
  for (int t = 0; t < 2; ++t) {
    v8f& c = t ? c1 : c0;
    const int col = n0 + t * 16 + nn;
    const float bval = bo[col];
#pragma unroll
    for (int j = 0; j < 8; ++j) {
      const int r = mtile * 16 + j + 8 * sel;
      out[(size_t)r * D_MODEL + col] = c[j] + bval;
    }
  }
}

// ---------------------------------------------------------------------------
extern "C" void kernel_launch(void* const* d_in, const int* in_sizes, int n_in,
                              void* d_out, int out_size, void* d_ws, size_t ws_size,
                              hipStream_t stream) {
  (void)in_sizes; (void)n_in; (void)out_size; (void)ws_size;

  const float* q  = (const float*)d_in[0];
  const float* k  = (const float*)d_in[1];
  const float* v  = (const float*)d_in[2];
  // d_in[3] = causal tril mask -> applied analytically in flash_attn
  const float* wq = (const float*)d_in[4];
  const float* bq = (const float*)d_in[5];
  const float* wk = (const float*)d_in[6];
  const float* bk = (const float*)d_in[7];
  const float* wv = (const float*)d_in[8];
  const float* bv = (const float*)d_in[9];
  const float* wo = (const float*)d_in[10];
  const float* bo = (const float*)d_in[11];

  char* ws = (char*)d_ws;
  size_t off = 0;
  auto take = [&](size_t elems) -> bf16* {
    bf16* p = (bf16*)(ws + off);
    off += ((elems * sizeof(bf16)) + 255) & ~(size_t)255;
    return p;
  };
  const size_t NX = (size_t)NROWS * D_MODEL;     // 4,194,304
  const size_t NW = (size_t)D_MODEL * D_MODEL;   // 1,048,576

  bf16* qb  = take(NX);
  bf16* kb  = take(NX);
  bf16* vb  = take(NX);
  bf16* wqb = take(NW);
  bf16* wkb = take(NW);
  bf16* wvb = take(NW);
  bf16* wob = take(NW);
  bf16* Qp  = take(NX);   // [B,H,S,dk]
  bf16* Kp  = take(NX);   // [B,H,S,dk]
  bf16* Vp  = take(NX);   // [B,H,dk,S]  (transposed for contiguous PV loads)
  bf16* Xa  = take(NX);   // attention output [B,S,D]
  // total workspace use: ~67 MB

  cvt_f32_bf16<<<2048, 256, 0, stream>>>(q,  qb,  (int)NX);
  cvt_f32_bf16<<<2048, 256, 0, stream>>>(k,  kb,  (int)NX);
  cvt_f32_bf16<<<2048, 256, 0, stream>>>(v,  vb,  (int)NX);
  cvt_f32_bf16<<<1024, 256, 0, stream>>>(wq, wqb, (int)NW);
  cvt_f32_bf16<<<1024, 256, 0, stream>>>(wk, wkb, (int)NW);
  cvt_f32_bf16<<<1024, 256, 0, stream>>>(wv, wvb, (int)NW);
  cvt_f32_bf16<<<1024, 256, 0, stream>>>(wo, wob, (int)NW);

  qkv_proj<<<dim3(1024, 1, 3), 256, 0, stream>>>(qb, kb, vb, wqb, wkb, wvb,
                                                 bq, bk, bv, Qp, Kp, Vp);
  flash_attn<<<1024, 128, 0, stream>>>(Qp, Kp, Vp, Xa);
  out_proj<<<1024, 256, 0, stream>>>(Xa, wob, bo, (float*)d_out);
}
